// RecurrentNet_52845277610211
// MI455X (gfx1250) — matrix operand order repaired and checked by reading, not compiled, not run on hardware
//
#include <hip/hip_runtime.h>
#include <hip/hip_bf16.h>

// Problem constants (match reference)
#define B_ 128
#define S_ 2048
#define D_ 128
#define H_ 128
#define C_ 10
#define G_ 384   // 3*H (r,z,n gate order)
#define K_ 128   // contraction dim for every GEMM here (D == H == 128)

typedef __attribute__((ext_vector_type(16))) __bf16 bf16x16;
typedef __attribute__((ext_vector_type(8)))  float  f32x8;

union FragU { uint4 q[2]; bf16x16 v; };

static __device__ __forceinline__ unsigned short f32_to_bf16(float f) {
  unsigned u = __float_as_uint(f);
  unsigned r = 0x7FFFu + ((u >> 16) & 1u);   // round-to-nearest-even
  return (unsigned short)((u + r) >> 16);
}

static __device__ __forceinline__ float sigmoid_f(float x) {
  return 1.0f / (1.0f + __expf(-x));
}

// Async DMA: global -> LDS, 16 bytes, tracked by ASYNCcnt (CDNA5).
// lds_off is a byte offset within the wave's LDS allocation.
static __device__ __forceinline__ void async_load_b128(unsigned lds_off, const void* gptr) {
  asm volatile("global_load_async_to_lds_b128 %0, %1, off"
               :: "v"(lds_off), "v"(gptr)
               : "memory");
}
static __device__ __forceinline__ void wait_asynccnt0() {
  asm volatile("s_wait_asynccnt 0x0" ::: "memory");
}

// ---------------------------------------------------------------------------
// f32 -> bf16 bulk convert, 8 elements / thread (b128 in, b128 out)
// ---------------------------------------------------------------------------
__global__ __launch_bounds__(256) void cvt_f32_to_bf16(const float* __restrict__ in,
                                                       unsigned short* __restrict__ out,
                                                       long n8) {
  long i = (long)blockIdx.x * blockDim.x + threadIdx.x;
  if (i >= n8) return;
  const float4* p = (const float4*)in + i * 2;
  float4 a = p[0], b = p[1];
  union { unsigned short h[8]; uint4 q; } u;
  u.h[0] = f32_to_bf16(a.x); u.h[1] = f32_to_bf16(a.y);
  u.h[2] = f32_to_bf16(a.z); u.h[3] = f32_to_bf16(a.w);
  u.h[4] = f32_to_bf16(b.x); u.h[5] = f32_to_bf16(b.y);
  u.h[6] = f32_to_bf16(b.z); u.h[7] = f32_to_bf16(b.w);
  ((uint4*)out)[i] = u.q;
}

// ---------------------------------------------------------------------------
// Input projection GEMM:  xp[s][b][g] = bias[g] + sum_k A[m][k] * W[g][k]
//   A: bf16 [M=B*S rows (m = b*S + s)][128], row-contiguous
//   W: bf16 [384][128]  (PyTorch W_ih layout, row n contiguous in k)
//   out: f32 [S][B][384]  (step-contiguous for the recurrence)
// Block = 8 waves. W staged once in LDS; each wave pins 3 n-tiles of B-frags
// in VGPRs (96 VGPRs) and streams 16 m-tiles of A from global (b128 loads).
// ---------------------------------------------------------------------------
__global__ __launch_bounds__(256) void gemm_inproj(const unsigned short* __restrict__ A,
                                                   const unsigned short* __restrict__ W,
                                                   const float* __restrict__ bias,
                                                   float* __restrict__ xp) {
  extern __shared__ char smem[];
  unsigned short* wlds = (unsigned short*)smem;   // [384][128] bf16 = 96 KB

  for (int i = threadIdx.x; i < G_ * K_ / 8; i += 256)
    ((uint4*)wlds)[i] = ((const uint4*)W)[i];
  __syncthreads();

  const int wv = threadIdx.x >> 5;
  const int l  = threadIdx.x & 31;
  const int lm = l & 15, lh = l >> 4;

  // B fragments: lane = column n, half -> K+8; two b128 at K and K+16 (matches A).
  bf16x16 Bf[3][4];
  float bv[3];
  for (int nt = 0; nt < 3; ++nt) {
    const int n0 = (wv + nt * 8) * 16;
    bv[nt] = bias[n0 + lm];
    for (int ks = 0; ks < 4; ++ks) {
      FragU u;
      const uint4* p = (const uint4*)&wlds[(n0 + lm) * K_ + ks * 32 + 8 * lh];
      u.q[0] = p[0]; u.q[1] = p[2];
      Bf[nt][ks] = u.v;
    }
  }

  const long mbase = (long)blockIdx.x * (16 * 16);
  for (int mt = 0; mt < 16; ++mt) {
    const long m0 = mbase + mt * 16;
    const unsigned short* arow = A + (m0 + lm) * K_;
    if (mt + 1 < 16) __builtin_prefetch(arow + 16 * K_, 0, 1);  // global_prefetch_b8

    // A fragments: lane = row m, half -> K+8; two contiguous b128 per k-step.
    bf16x16 Af[4];
#pragma unroll
    for (int ks = 0; ks < 4; ++ks) {
      FragU u;
      u.q[0] = *(const uint4*)(arow + ks * 32 + 8 * lh);
      u.q[1] = *(const uint4*)(arow + ks * 32 + 8 * lh + 16);
      Af[ks] = u.v;
    }

#pragma unroll
    for (int nt = 0; nt < 3; ++nt) {
      f32x8 c = {0, 0, 0, 0, 0, 0, 0, 0};
#pragma unroll
      for (int ks = 0; ks < 4; ++ks)
        c = __builtin_amdgcn_wmma_f32_16x16x32_bf16(false, Af[ks], false, Bf[nt][ks],
                                                    (short)0, c, false, false);
      const int g = (wv + nt * 8) * 16 + lm;
#pragma unroll
      for (int r = 0; r < 8; ++r) {
        const long m = m0 + r + 8 * lh;
        const long s = m & (S_ - 1);   // S = 2048
        const long b = m >> 11;
        xp[(s * B_ + b) * G_ + g] = c[r] + bv[nt];
      }
    }
  }
}

// ---------------------------------------------------------------------------
// Persistent GRU recurrence. One block per 16 batch rows; h kept in LDS,
// W_hh B-fragments pinned in VGPRs for all 2048 steps. The next step's xp
// slice (16x384 f32 = 24 KB) is DMA'd global->LDS asynchronously (ASYNCcnt)
// while the current step's gate math + next step's WMMA run, removing HBM latency from
// the 2048-long serial chain.
//   xp : f32 [S][B][384] precomputed input projections (+b_ih)
//   Whh: bf16 [384][128]
//   ybf: bf16 [B][S][128] layer output (nullptr for last layer)
// ---------------------------------------------------------------------------
#define WLDS_OFF  0
#define HBF_OFF   (G_*K_*2)                          // 98304
#define HF_OFF    (HBF_OFF + 16*H_*2)                // +4096
#define HP_OFF    (HF_OFF + 16*H_*4)                 // +8192
#define XBUF_OFF  (HP_OFF + 16*G_*4)                 // +24576 -> 135168
#define XSLICE    (16*G_*4)                          // 24576 bytes per step slice
#define REC_SMEM  (XBUF_OFF + 2*XSLICE)              // 184320 bytes

__global__ __launch_bounds__(256) void gru_rec(const float* __restrict__ xp,
                                               const unsigned short* __restrict__ Whh,
                                               const float* __restrict__ bhh,
                                               unsigned short* __restrict__ ybf,
                                               float* __restrict__ hlast) {
  extern __shared__ char smem[];
  unsigned short* wlds = (unsigned short*)(smem + WLDS_OFF);  // [384][128] bf16
  unsigned short* hbf  = (unsigned short*)(smem + HBF_OFF);   // [16][128] bf16 h
  float* hf   = (float*)(smem + HF_OFF);                      // [16][128] f32 h
  float* hp   = (float*)(smem + HP_OFF);                      // [16][384] f32 h@Whh^T+bhh
  float* xbuf = (float*)(smem + XBUF_OFF);                    // double-buffered xp slice
  const unsigned xbuf_lds = (unsigned)__builtin_amdgcn_groupstaticsize() + XBUF_OFF;

  const int tid = threadIdx.x;
  const int b0 = blockIdx.x * 16;

  // Kick off DMA of step 0's xp slice immediately.
  {
    const char* g0 = (const char*)(xp + (long)b0 * G_);
    const unsigned dst0 = xbuf_lds;  // buffer parity 0
#pragma unroll
    for (int i = 0; i < 6; ++i) {
      const unsigned off = (unsigned)tid * 16u + (unsigned)i * 4096u;
      async_load_b128(dst0 + off, g0 + off);
    }
  }

  for (int i = tid; i < G_ * K_ / 8; i += 256) ((uint4*)wlds)[i] = ((const uint4*)Whh)[i];
  for (int i = tid; i < 16 * H_ / 8; i += 256) ((uint4*)hbf)[i] = make_uint4(0, 0, 0, 0);
  for (int i = tid; i < 16 * H_ / 4; i += 256) ((float4*)hf)[i] = make_float4(0.f, 0.f, 0.f, 0.f);
  __syncthreads();

  const int wv = tid >> 5, l = tid & 31, lm = l & 15, lh = l >> 4;

  bf16x16 Bf[3][4];
  float bhv[3];
  for (int nt = 0; nt < 3; ++nt) {
    const int n0 = (wv + nt * 8) * 16;
    bhv[nt] = bhh[n0 + lm];
    for (int ks = 0; ks < 4; ++ks) {
      FragU u;
      const uint4* p = (const uint4*)&wlds[(n0 + lm) * K_ + ks * 32 + 8 * lh];
      u.q[0] = p[0]; u.q[1] = p[2];
      Bf[nt][ks] = u.v;
    }
  }

  for (int t = 0; t < S_; ++t) {
    // --- hp = h @ Whh^T + bhh (WMMA over the 16x128 h tile) ---
    bf16x16 Af[4];
#pragma unroll
    for (int ks = 0; ks < 4; ++ks) {
      FragU u;
      u.q[0] = *(const uint4*)&hbf[lm * H_ + ks * 32 + 8 * lh];
      u.q[1] = *(const uint4*)&hbf[lm * H_ + ks * 32 + 8 * lh + 16];
      Af[ks] = u.v;
    }
#pragma unroll
    for (int nt = 0; nt < 3; ++nt) {
      f32x8 c = {0, 0, 0, 0, 0, 0, 0, 0};
#pragma unroll
      for (int ks = 0; ks < 4; ++ks)
        c = __builtin_amdgcn_wmma_f32_16x16x32_bf16(false, Af[ks], false, Bf[nt][ks],
                                                    (short)0, c, false, false);
      const int g = (wv + nt * 8) * 16 + lm;
#pragma unroll
      for (int r = 0; r < 8; ++r)
        hp[(r + 8 * lh) * G_ + g] = c[r] + bhv[nt];
    }

    // Step t's xp DMA (issued at t-1) must be complete, then publish hp+xbuf.
    wait_asynccnt0();
    __syncthreads();

    // Issue DMA for step t+1 into the other buffer; overlaps the gate phase
    // below plus the next step's WMMA phase.
    if (t + 1 < S_) {
      const char* gnext = (const char*)(xp + ((long)(t + 1) * B_ + b0) * G_);
      const unsigned dstn = xbuf_lds + (unsigned)((t + 1) & 1) * XSLICE;
#pragma unroll
      for (int i = 0; i < 6; ++i) {
        const unsigned off = (unsigned)tid * 16u + (unsigned)i * 4096u;
        async_load_b128(dstn + off, gnext + off);
      }
    }

    // --- fused gates: r,z sigmoid; n tanh; h = (1-z)*n + z*h ---
    const float* xpt = xbuf + (t & 1) * (16 * G_);
#pragma unroll
    for (int j = 0; j < 4; ++j) {
      const int idx = tid * 4 + j;        // 1024 pairs = 2048 elems = 16x128
      const int row = idx >> 6;
      const int col = (idx & 63) << 1;
      const float* xr = xpt + row * G_ + col;
      float2 xrv = *(const float2*)xr;
      float2 xzv = *(const float2*)(xr + H_);
      float2 xnv = *(const float2*)(xr + 2 * H_);
      const float* hpr = hp + row * G_ + col;
      float2 hrv = *(const float2*)hpr;
      float2 hzv = *(const float2*)(hpr + H_);
      float2 hnv = *(const float2*)(hpr + 2 * H_);
      float2 hov = *(const float2*)&hf[row * H_ + col];
      float r0 = sigmoid_f(xrv.x + hrv.x);
      float r1 = sigmoid_f(xrv.y + hrv.y);
      float z0 = sigmoid_f(xzv.x + hzv.x);
      float z1 = sigmoid_f(xzv.y + hzv.y);
      float n0v = tanhf(xnv.x + r0 * hnv.x);
      float n1v = tanhf(xnv.y + r1 * hnv.y);
      float h0 = (1.f - z0) * n0v + z0 * hov.x;
      float h1 = (1.f - z1) * n1v + z1 * hov.y;
      *(float2*)&hf[row * H_ + col] = make_float2(h0, h1);
      unsigned pk = (unsigned)f32_to_bf16(h0) | ((unsigned)f32_to_bf16(h1) << 16);
      *(unsigned*)&hbf[row * H_ + col] = pk;
      if (ybf)
        *(unsigned*)&ybf[((long)(b0 + row) * S_ + t) * H_ + col] = pk;
    }
    __syncthreads();
  }

  for (int i = tid; i < 16 * H_; i += 256) {
    const int row = i >> 7, col = i & (H_ - 1);
    hlast[(long)(b0 + row) * H_ + col] = hf[i];
  }
}

// ---------------------------------------------------------------------------
// Output head: out[b][c] = b_out[c] + h_last[b] . W_out[c]   (128x10, tiny)
// ---------------------------------------------------------------------------
__global__ __launch_bounds__(256) void out_linear(const float* __restrict__ hl,
                                                  const float* __restrict__ Wo,
                                                  const float* __restrict__ bo,
                                                  float* __restrict__ out) {
  const int tid = blockIdx.x * blockDim.x + threadIdx.x;
  if (tid >= B_ * C_) return;
  const int b = tid / C_, c = tid % C_;
  const float* hr = hl + (long)b * H_;
  const float* wr = Wo + (long)c * H_;
  float acc = bo[c];
#pragma unroll 8
  for (int k = 0; k < H_; ++k) acc = fmaf(hr[k], wr[k], acc);
  out[tid] = acc;
}

// ---------------------------------------------------------------------------
extern "C" void kernel_launch(void* const* d_in, const int* in_sizes, int n_in,
                              void* d_out, int out_size, void* d_ws, size_t ws_size,
                              hipStream_t stream) {
  (void)in_sizes; (void)n_in; (void)out_size; (void)ws_size;
  const float* x    = (const float*)d_in[0];
  const float* Wih0 = (const float*)d_in[1];
  const float* Whh0 = (const float*)d_in[2];
  const float* bih0 = (const float*)d_in[3];
  const float* bhh0 = (const float*)d_in[4];
  const float* Wih1 = (const float*)d_in[5];
  const float* Whh1 = (const float*)d_in[6];
  const float* bih1 = (const float*)d_in[7];
  const float* bhh1 = (const float*)d_in[8];
  const float* Wout = (const float*)d_in[9];
  const float* bout = (const float*)d_in[10];
  float* out = (float*)d_out;

  // Workspace partition (~537 MB): xbf 64M, ybf 64M, xp 384M, weights, h_last
  char* ws = (char*)d_ws;
  size_t off = 0;
  auto take = [&](size_t bytes) -> char* {
    char* p = ws + off;
    off += (bytes + 255) & ~(size_t)255;
    return p;
  };
  unsigned short* xbf = (unsigned short*)take((size_t)B_ * S_ * D_ * 2);
  unsigned short* ybf = (unsigned short*)take((size_t)B_ * S_ * H_ * 2);
  float*          xp  = (float*)take((size_t)S_ * B_ * G_ * 4);
  unsigned short* wi0 = (unsigned short*)take((size_t)G_ * D_ * 2);
  unsigned short* wh0 = (unsigned short*)take((size_t)G_ * H_ * 2);
  unsigned short* wi1 = (unsigned short*)take((size_t)G_ * H_ * 2);
  unsigned short* wh1 = (unsigned short*)take((size_t)G_ * H_ * 2);
  float*        hlast = (float*)take((size_t)B_ * H_ * 4);

  // bf16 conversions (activations + all matmul weights)
  const long n8x = (long)B_ * S_ * D_ / 8;
  cvt_f32_to_bf16<<<(unsigned)((n8x + 255) / 256), 256, 0, stream>>>(x, xbf, n8x);
  const long n8w = (long)G_ * K_ / 8;
  cvt_f32_to_bf16<<<(unsigned)((n8w + 255) / 256), 256, 0, stream>>>(Wih0, wi0, n8w);
  cvt_f32_to_bf16<<<(unsigned)((n8w + 255) / 256), 256, 0, stream>>>(Whh0, wh0, n8w);
  cvt_f32_to_bf16<<<(unsigned)((n8w + 255) / 256), 256, 0, stream>>>(Wih1, wi1, n8w);
  cvt_f32_to_bf16<<<(unsigned)((n8w + 255) / 256), 256, 0, stream>>>(Whh1, wh1, n8w);

  const int gemm_blocks = (B_ * S_) / (16 * 16);   // 1024
  const size_t gemm_smem = (size_t)G_ * K_ * 2;    // 96 KB

  // Layer 0: input projection GEMM, then persistent recurrence (emits y0 bf16)
  gemm_inproj<<<gemm_blocks, 256, gemm_smem, stream>>>(xbf, wi0, bih0, xp);
  gru_rec<<<B_ / 16, 256, REC_SMEM, stream>>>(xp, wh0, bhh0, ybf, hlast);

  // Layer 1: same, consuming y0; no y output, only h_last
  gemm_inproj<<<gemm_blocks, 256, gemm_smem, stream>>>(ybf, wi1, bih1, xp);
  gru_rec<<<B_ / 16, 256, REC_SMEM, stream>>>(xp, wh1, bhh1, nullptr, hlast);

  // Output head
  out_linear<<<(B_ * C_ + 255) / 256, 256, 0, stream>>>(hlast, Wout, bout, out);
}
